// MultiheadBlockAttention_4715874091537
// MI455X (gfx1250) — matrix operand order repaired and checked by reading, not compiled, not run on hardware
//
#include <hip/hip_runtime.h>
#include <hip/hip_bf16.h>

// CDNA5 (gfx1250) wave32 WMMA types
typedef __attribute__((ext_vector_type(16))) _Float16 v16h;
typedef __attribute__((ext_vector_type(8)))  float    v8f;

// ---------------------------------------------------------------------------
// Fragment packing layouts (K = D = 512 fixed):
//   A-fragment (16x32 f16 tile): pA[(tr*16 + tk)*512 + lane*16 + e]
//     value = A[(tr*16 + (lane&15))*512 + tk*32 + akk(e, lane>>4)]
//     akk(e,g) = ((e>>3)<<4) + (((e>>1)&3)<<1) + (e&1) + (g<<3)
//   B-fragment (32x16 f16 tile): pW[(tk*32 + tc)*512 + lane*16 + e]
//     value = W[(tk*32 + ((e>>3)<<4) + ((lane>>4)<<3) + (e&7))*512 + tc*16 + (lane&15)]
// Each lane's 16 f16 are contiguous (32 B) -> fragment load = 2x b128.
// ---------------------------------------------------------------------------

// Pack one 512x512 f32 weight matrix into B-fragment f16 order. One wave per
// 32x16 tile; 8 waves/block; grid = 512/8 = 64 blocks. One-time 1 MB pass.
__global__ __launch_bounds__(256) void pack_w_f16(
    const float* __restrict__ W, _Float16* __restrict__ pW)
{
  const unsigned lane = threadIdx.x & 31u, wid = threadIdx.x >> 5;
  const unsigned tile = blockIdx.x * 8u + wid;   // 0..511 = tk*32 + tc
  const unsigned tk = tile >> 5, tc = tile & 31u;
  const unsigned r = lane & 15u, g = lane >> 4;
  v16h v;
#pragma unroll
  for (int e = 0; e < 16; ++e) {
    unsigned k = tk * 32u + ((unsigned)(e >> 3) << 4) + (g << 3) + ((unsigned)e & 7u);
    v[e] = (_Float16)W[(size_t)k * 512u + tc * 16u + r];
  }
  *(v16h*)(pW + (size_t)tile * 512u + lane * 16u) = v;
}

// Pack activations [rows x 512] f32 into A-fragment f16 order.
// grid.x = rows/16 (row-tile); 8 waves/block, wave handles 2 k-tiles.
__global__ __launch_bounds__(256) void pack_a_f16(
    const float* __restrict__ A, _Float16* __restrict__ pA)
{
  const unsigned lane = threadIdx.x & 31u, wid = threadIdx.x >> 5;
  const unsigned tr = blockIdx.x;
  const unsigned r = lane & 15u, g = lane >> 4;
#pragma unroll
  for (unsigned q = 0; q < 2u; ++q) {
    const unsigned tk = wid * 2u + q;
    const float* src = A + (size_t)(tr * 16u + r) * 512u + tk * 32u;
    v16h v;
#pragma unroll
    for (int e = 0; e < 16; ++e) {
      unsigned kk = ((unsigned)(e >> 3) << 4)
                  + (((unsigned)(e >> 1) & 3u) << 1)
                  + (unsigned)(e & 1)
                  + (g << 3);
      v[e] = (_Float16)src[kk];
    }
    *(v16h*)(pA + (size_t)(tr * 16u + tk) * 512u + lane * 16u) = v;
  }
}

// ---------------------------------------------------------------------------
// GEMM: C[rows x 512] = unpack(pA) @ unpack(pW) + bias, f32 accumulate/store.
// Block = 256 threads (8 waves). Each wave owns a 32-row x 64-col output
// macro-tile: 2 A fragments x 4 B fragments -> 8 independent accumulators.
// All 6 fragment loads of a k-step are issued into distinct registers before
// the 8 WMMAs, so loadcnt waits stagger instead of serializing each WMMA.
// blockIdx.x = 32-row strip; rows must be a multiple of 32.
// ---------------------------------------------------------------------------
__global__ __launch_bounds__(256) void gemm512_f16wmma(
    const _Float16* __restrict__ pA, const _Float16* __restrict__ pW,
    const float* __restrict__ bias, float* __restrict__ C)
{
  const unsigned lane = threadIdx.x & 31u, wid = threadIdx.x >> 5;
  const unsigned g = lane >> 4, r = lane & 15u;
  const unsigned tr0 = blockIdx.x * 2u;      // first of two 16-row tiles

  v8f c[2][4];
#pragma unroll
  for (int i = 0; i < 2; ++i)
#pragma unroll
    for (int j = 0; j < 4; ++j)
      c[i][j] = (v8f){0.f,0.f,0.f,0.f,0.f,0.f,0.f,0.f};

#pragma unroll 2
  for (unsigned tk = 0; tk < 16u; ++tk) {
    const _Float16* abase = pA + (size_t)(tr0 * 16u + tk) * 512u + lane * 16u;
    const v16h a0 = *(const v16h*)(abase);
    const v16h a1 = *(const v16h*)(abase + 16u * 512u);   // tile (tr0+1, tk)
    v16h b[4];
#pragma unroll
    for (int j = 0; j < 4; ++j) {
      const unsigned tc = wid * 4u + (unsigned)j;
      b[j] = *(const v16h*)(pW + (size_t)(tk * 32u + tc) * 512u + lane * 16u);
    }
#pragma unroll
    for (int j = 0; j < 4; ++j) {
      c[0][j] = __builtin_amdgcn_wmma_f32_16x16x32_f16(
          false, a0, false, b[j], (short)0, c[0][j], false, false);
      c[1][j] = __builtin_amdgcn_wmma_f32_16x16x32_f16(
          false, a1, false, b[j], (short)0, c[1][j], false, false);
    }
  }

  // C layout: VGPR r8 -> row r8 (lanes 0-15) / r8+8 (lanes 16-31), col = lane&15.
#pragma unroll
  for (int i = 0; i < 2; ++i) {
    const unsigned row0 = (tr0 + (unsigned)i) * 16u;
#pragma unroll
    for (int j = 0; j < 4; ++j) {
      const unsigned col = (wid * 4u + (unsigned)j) * 16u + r;
      const float bb = bias[col];
#pragma unroll
      for (int r8 = 0; r8 < 8; ++r8) {
        const unsigned row = row0 + (unsigned)r8 + (g << 3);
        C[(size_t)row * 512u + col] = c[i][j][r8] + bb;
      }
    }
  }
}

// ---------------------------------------------------------------------------
// Temporal attention: one 64-thread block per (b,h,s).
// Q[8x64] x K[24x64]^T -> mask -> softmax -> w_t out; w @ V[24x64] -> qkv_t.
// qkv_t written in [B,S,M,D] layout with head h at cols h*64.. so the Wo GEMM
// consumes it directly (matches reference transpose(0,2,3,1,4).reshape).
// ---------------------------------------------------------------------------
__global__ __launch_bounds__(64) void attn_temporal(
    const float* __restrict__ Qt, const float* __restrict__ Kt,
    const float* __restrict__ Vt, const float* __restrict__ Ks,
    const float* __restrict__ Vs,
    const int* __restrict__ mask_t, const int* __restrict__ mask_s,
    float* __restrict__ w_t_out, float* __restrict__ qkvt)
{
  __shared__ float sQ[8][64];
  __shared__ float sK[24][64];
  __shared__ float sV[24][64];
  __shared__ float sL[8][24];

  const unsigned bid = blockIdx.x;
  const unsigned b = bid >> 12, h = (bid >> 9) & 7u, s = bid & 511u;
  const unsigned t = threadIdx.x;
  const float NEG_INF = -__builtin_inff();

  for (unsigned i = t; i < 512u; i += 64u) {
    unsigned m = i >> 6, d = i & 63u;
    sQ[m][d] = Qt[(size_t)((b * 512u + s) * 8u + m) * 512u + h * 64u + d];
  }
  for (unsigned i = t; i < 1536u; i += 64u) {
    unsigned n = i >> 6, d = i & 63u;
    if (n < 8u) {
      size_t base = (size_t)((b * 512u + s) * 8u + n) * 512u + h * 64u + d;
      sK[n][d] = Kt[base]; sV[n][d] = Vt[base];
    } else {
      size_t base = (size_t)(b * 16u + (n - 8u)) * 512u + h * 64u + d;
      sK[n][d] = Ks[base]; sV[n][d] = Vs[base];
    }
  }
  __syncthreads();

  for (unsigned idx = t; idx < 192u; idx += 64u) {
    unsigned m = idx / 24u, n = idx % 24u;
    float acc = 0.f;
#pragma unroll 8
    for (int d = 0; d < 64; ++d) acc += sQ[m][d] * sK[n][d];
    int mk = (n < 8u) ? mask_t[(b * 512u + s) * 8u + n]
                      : mask_s[b * 16u + (n - 8u)];
    sL[m][n] = acc * 0.125f + (mk ? 0.f : NEG_INF);
  }
  __syncthreads();

  if (t < 8u) {
    const unsigned m = t;
    float mx = NEG_INF;
    for (int n = 0; n < 24; ++n) mx = fmaxf(mx, sL[m][n]);
    float e[24], sum = 0.f;
    for (int n = 0; n < 24; ++n) { e[n] = __expf(sL[m][n] - mx); sum += e[n]; }
    const float inv = 1.f / sum;
    const size_t wb = (size_t)(((b * 8u + h) * 512u + s) * 8u + m) * 24u;
    for (int n = 0; n < 24; ++n) {
      float w = e[n] * inv;
      sL[m][n] = w;
      w_t_out[wb + n] = w;
    }
  }
  __syncthreads();

  for (unsigned f = t; f < 512u; f += 64u) {
    unsigned m = f >> 6, d = f & 63u;
    float acc = 0.f;
#pragma unroll
    for (int n = 0; n < 24; ++n) acc += sL[m][n] * sV[n][d];
    qkvt[(size_t)((b * 512u + s) * 8u + m) * 512u + h * 64u + d] = acc;
  }
}

// ---------------------------------------------------------------------------
// Static attention: one 256-thread block per (b,h,qi). 4112 keys
// (4096 temporal flat + 16 static). Two-pass softmax using the w_s output
// buffer as logit storage; LDS tree reductions; 4-way chunked weighted-V.
// ---------------------------------------------------------------------------
__global__ __launch_bounds__(256) void attn_static(
    const float* __restrict__ Qs, const float* __restrict__ Kt,
    const float* __restrict__ Vt, const float* __restrict__ Ks,
    const float* __restrict__ Vs,
    const int* __restrict__ mask_t, const int* __restrict__ mask_s,
    float* __restrict__ w_s_out, float* __restrict__ qkvs)
{
  __shared__ float sq[64];
  __shared__ float red[256];
  __shared__ float accS[4][64];

  const unsigned bid = blockIdx.x;
  const unsigned b = bid >> 7, h = (bid >> 4) & 7u, qi = bid & 15u;
  const unsigned t = threadIdx.x;
  const float NEG_INF = -__builtin_inff();

  if (t < 64u) sq[t] = Qs[(size_t)(b * 16u + qi) * 512u + h * 64u + t];
  __syncthreads();

  const size_t wb = (size_t)((b * 8u + h) * 16u + qi) * 4112u;

  // pass 1: logits + running max
  float lmax = NEG_INF;
  for (unsigned n = t; n < 4112u; n += 256u) {
    const float* krow;
    int mk;
    if (n < 4096u) {
      krow = Kt + (size_t)(b * 4096u + n) * 512u + h * 64u;
      mk = mask_t[b * 4096u + n];
    } else {
      krow = Ks + (size_t)(b * 16u + (n - 4096u)) * 512u + h * 64u;
      mk = mask_s[b * 16u + (n - 4096u)];
    }
    float acc = 0.f;
#pragma unroll 8
    for (int d = 0; d < 64; ++d) acc += sq[d] * krow[d];
    float l = acc * 0.125f + (mk ? 0.f : NEG_INF);
    w_s_out[wb + n] = l;
    lmax = fmaxf(lmax, l);
  }
  red[t] = lmax; __syncthreads();
  for (unsigned off = 128u; off; off >>= 1) {
    if (t < off) red[t] = fmaxf(red[t], red[t + off]);
    __syncthreads();
  }
  const float rmax = red[0];
  __syncthreads();

  // pass 2: exp + sum (each thread re-reads only its own writes)
  float lsum = 0.f;
  for (unsigned n = t; n < 4112u; n += 256u) {
    float e = __expf(w_s_out[wb + n] - rmax);
    w_s_out[wb + n] = e;
    lsum += e;
  }
  red[t] = lsum; __syncthreads();
  for (unsigned off = 128u; off; off >>= 1) {
    if (t < off) red[t] += red[t + off];
    __syncthreads();
  }
  const float inv = 1.f / red[0];
  __syncthreads();

  for (unsigned n = t; n < 4112u; n += 256u) w_s_out[wb + n] *= inv;
  __syncthreads();

  // weighted V: thread -> (chunk c4 of 1028 keys, dim d)
  const unsigned d = t & 63u, c4 = t >> 6;
  float acc = 0.f;
  for (unsigned n = c4 * 1028u; n < c4 * 1028u + 1028u; ++n) {
    float w = w_s_out[wb + n];
    float v = (n < 4096u)
        ? Vt[(size_t)(b * 4096u + n) * 512u + h * 64u + d]
        : Vs[(size_t)(b * 16u + (n - 4096u)) * 512u + h * 64u + d];
    acc += w * v;
  }
  accS[c4][d] = acc;
  __syncthreads();
  if (t < 64u) {
    qkvs[(size_t)(b * 16u + qi) * 512u + h * 64u + t] =
        accS[0][t] + accS[1][t] + accS[2][t] + accS[3][t];
  }
}

// ---------------------------------------------------------------------------
// Launch
// ---------------------------------------------------------------------------
extern "C" void kernel_launch(void* const* d_in, const int* in_sizes, int n_in,
                              void* d_out, int out_size, void* d_ws, size_t ws_size,
                              hipStream_t stream) {
  const float* q_t = (const float*)d_in[0];
  const float* k_t = (const float*)d_in[1];
  const float* v_t = (const float*)d_in[2];
  const float* q_s = (const float*)d_in[3];
  const float* k_s = (const float*)d_in[4];
  const float* v_s = (const float*)d_in[5];
  const int*   m_t = (const int*)d_in[6];
  const int*   m_s = (const int*)d_in[7];
  const float* Wq = (const float*)d_in[8];  const float* bq = (const float*)d_in[9];
  const float* Wk = (const float*)d_in[10]; const float* bk = (const float*)d_in[11];
  const float* Wv = (const float*)d_in[12]; const float* bv = (const float*)d_in[13];
  const float* Wo = (const float*)d_in[14]; const float* bo = (const float*)d_in[15];

  float* out = (float*)d_out;
  // output regions (flat f32, in reference return order)
  const size_t O_OUT_T = 0;                       // [8,512,8,512]  = 16,777,216
  const size_t O_W_T   = 16777216;                // [8,8,512,8,24] =  6,291,456
  const size_t O_OUT_S = 23068672;                // [8,16,512]     =     65,536
  const size_t O_W_S   = 23134208;                // [8,8,16,4112]  =  4,210,688

  // workspace carve-up; ~304 MB total
  float* ws   = (float*)d_ws;
  const size_t NT = 16777216;  // 32768 x 512
  const size_t NS = 65536;     // 128 x 512
  float* Qt   = ws;
  float* Kt   = Qt + NT;
  float* Vt   = Kt + NT;
  float* Qs   = Vt + NT;
  float* Ks   = Qs + NS;
  float* Vs   = Ks + NS;
  float* qkvt = Vs + NS;
  float* qkvs = qkvt + NT;
  _Float16* pWq = (_Float16*)(qkvs + NS);          // packed weights, 256K halves each
  _Float16* pWk = pWq + 262144;
  _Float16* pWv = pWk + 262144;
  _Float16* pWo = pWv + 262144;
  _Float16* pA  = pWo + 262144;                    // packed-A scratch, 16,777,216 halves

  const dim3 blk(256);

  // one-time weight packing (f32 -> f16 B-fragment layout, 512 KB each)
  pack_w_f16<<<64, blk, 0, stream>>>(Wq, pWq);
  pack_w_f16<<<64, blk, 0, stream>>>(Wk, pWk);
  pack_w_f16<<<64, blk, 0, stream>>>(Wv, pWv);
  pack_w_f16<<<64, blk, 0, stream>>>(Wo, pWo);

  // projections: pack A then GEMM (pA scratch reused sequentially)
  pack_a_f16<<<2048, blk, 0, stream>>>(q_t, pA);
  gemm512_f16wmma<<<1024, blk, 0, stream>>>(pA, pWq, bq, Qt);
  pack_a_f16<<<2048, blk, 0, stream>>>(k_t, pA);
  gemm512_f16wmma<<<1024, blk, 0, stream>>>(pA, pWk, bk, Kt);
  pack_a_f16<<<2048, blk, 0, stream>>>(v_t, pA);
  gemm512_f16wmma<<<1024, blk, 0, stream>>>(pA, pWv, bv, Vt);
  pack_a_f16<<<8, blk, 0, stream>>>(q_s, pA);
  gemm512_f16wmma<<<4, blk, 0, stream>>>(pA, pWq, bq, Qs);
  pack_a_f16<<<8, blk, 0, stream>>>(k_s, pA);
  gemm512_f16wmma<<<4, blk, 0, stream>>>(pA, pWk, bk, Ks);
  pack_a_f16<<<8, blk, 0, stream>>>(v_s, pA);
  gemm512_f16wmma<<<4, blk, 0, stream>>>(pA, pWv, bv, Vs);

  // attention cores
  attn_temporal<<<32768, 64, 0, stream>>>(Qt, Kt, Vt, Ks, Vs, m_t, m_s,
                                          out + O_W_T, qkvt);
  attn_static<<<1024, 256, 0, stream>>>(Qs, Kt, Vt, Ks, Vs, m_t, m_s,
                                        out + O_W_S, qkvs);

  // output projections
  pack_a_f16<<<2048, blk, 0, stream>>>(qkvt, pA);
  gemm512_f16wmma<<<1024, blk, 0, stream>>>(pA, pWo, bo, out + O_OUT_T);
  pack_a_f16<<<8, blk, 0, stream>>>(qkvs, pA);
  gemm512_f16wmma<<<4, blk, 0, stream>>>(pA, pWo, bo, out + O_OUT_S);
}
